// SwinEncoderBlock_5892695130142
// MI455X (gfx1250) — compile-verified, matrix operand order checked by reading, not run on hardware
//
#include <hip/hip_runtime.h>

// ---------------- problem constants ----------------
#define SB   64           // batch
#define SHW  112          // H == W
#define SC   96           // channels
#define SWS  7            // window size
#define SSH  3            // shift
#define SNH  3            // heads
#define SN   49           // tokens per window
#define SHD  32           // head dim
#define SHID 384          // MLP hidden
#define SNW  256          // windows per image (16x16)
#define SWTOT (SB*SNW)    // 16384 windows
#define SM0  (SWTOT*SN)   // 802816 rows
#define SSCALE 0.17677669529663687f   // 32^-0.5

typedef unsigned int  u32;
typedef unsigned short u16;
typedef __attribute__((ext_vector_type(16))) __bf16 bf16x16;
typedef __attribute__((ext_vector_type(8)))  float  f32x8;
typedef __attribute__((ext_vector_type(4)))  unsigned int u32x4;
typedef __attribute__((ext_vector_type(8)))  int    i32x8;
typedef __attribute__((ext_vector_type(4)))  int    i32x4;

union Frag { bf16x16 v; u32 d[8]; u16 h[16]; };
union CAcc { f32x8 v; float f[8]; };

__device__ inline f32x8 zero8() {
  f32x8 z = {0.f,0.f,0.f,0.f,0.f,0.f,0.f,0.f};
  return z;
}

__device__ inline u16 f2bf(float f) {
  union { float f; u32 u; } x; x.f = f;
  u32 r = x.u + 0x7FFFu + ((x.u >> 16) & 1u);     // RNE
  return (u16)(r >> 16);
}

// dword v (0..7) of a 16-bit WMMA operand holds K pair starting at:
__device__ inline int kofs(int v, int hf) {
  return ((v & 3) << 1) + ((v >> 2) << 4) + (hf << 3);
}
// element e (0..15) of a 16-bit WMMA operand maps to K:
__device__ inline int kelem(int e, int hf) {
  return (e & 7) + ((e >> 3) << 4) + (hf << 3);
}

__device__ inline f32x8 wmma_bf16(const Frag& a, const Frag& b, f32x8 c) {
  return __builtin_amdgcn_wmma_f32_16x16x32_bf16(false, a.v, false, b.v,
                                                 (short)0, c, false, false);
}

// ---- TDM: DMA a [rows x cols] bf16 panel from global into LDS ----
// D# per CDNA5 ISA ch.8: group0 = {count/lds_addr/global_addr/type},
// group1 = {data_size, tensor dims, tile dims, stride}; 2-D tile = whole panel.
// This toolchain exposes the 6-arg builtin (extra int32x8 group, then cpol).
__device__ inline void tdm_load_panel(const u16* gsrc, u16* lds_dst,
                                      int rows, int cols) {
  u32x4 g0;
  i32x8 g1;
  i32x4 g2 = {0,0,0,0}, g3 = {0,0,0,0};
  i32x8 g4 = {0,0,0,0,0,0,0,0};
  unsigned long long ga = (unsigned long long)(uintptr_t)gsrc;
  g0[0] = 1u;                                          // D# count = 1
  g0[1] = (u32)(uintptr_t)lds_dst;                     // LDS byte address (low 32b of flat)
  g0[2] = (u32)ga;                                     // global_addr[31:0]
  g0[3] = (u32)((ga >> 32) & 0x01FFFFFFu) | (2u << 30);// global_addr[56:32] | type=2
  g1[0] = (1 << 16);                                   // data_size = 2 bytes
  g1[1] = (cols & 0xFFFF) << 16;                       // tensor_dim0[15:0]
  g1[2] = ((cols >> 16) & 0xFFFF) | ((rows & 0xFFFF) << 16); // dim0 hi | dim1 lo
  g1[3] = ((rows >> 16) & 0xFFFF) | ((cols & 0xFFFF) << 16); // dim1 hi | tile_dim0
  g1[4] = (rows & 0xFFFF);                             // tile_dim1 (tile_dim2=0)
  g1[5] = cols;                                        // tensor_dim0_stride[31:0]
  g1[6] = 0;
  g1[7] = 0;
  __builtin_amdgcn_tensor_load_to_lds(g0, g1, g2, g3, g4, 0);
  __builtin_amdgcn_s_wait_tensorcnt(0);
}

// ---------------- K0: weight transpose/convert + attn bias gather ----------------
__global__ void k_prep(const float* qkv_w, const float* proj_w,
                       const float* fc1_w, const float* fc2_w,
                       const float* rpb, const int* rel,
                       u16* qkv_wt, u16* proj_wt, u16* fc1_wt, u16* fc2_wt,
                       float* bias_attn) {
  int i = blockIdx.x * 256 + threadIdx.x;
  if (i < 288*96) { int n = i/96,  kk = i%96;  qkv_wt[i]  = f2bf(qkv_w[kk*288 + n]); }
  if (i < 96*96)  { int n = i/96,  kk = i%96;  proj_wt[i] = f2bf(proj_w[kk*96 + n]); }
  if (i < 384*96) { int n = i/96,  kk = i%96;  fc1_wt[i]  = f2bf(fc1_w[kk*384 + n]); }
  if (i < 96*384) { int n = i/384, kk = i%384; fc2_wt[i]  = f2bf(fc2_w[kk*96 + n]); }
  if (i < 3*49*49){ int h = i/2401, rem = i%2401; bias_attn[i] = rpb[rel[rem]*3 + h]; }
}

// ---------------- K1: LN1 + cyclic shift + window partition -> bf16 ----------------
__global__ __launch_bounds__(256) void k_ln1(const float* __restrict__ x,
                                             const float* __restrict__ g,
                                             const float* __restrict__ bt,
                                             u16* __restrict__ out) {
  int wv = threadIdx.x >> 5, lane = threadIdx.x & 31;
  int row = blockIdx.x * 8 + wv;                 // [0, 802816)
  int w = row / SN, n = row % SN;
  int bb = w >> 8, wi = w & 255, wy = wi >> 4, wx = wi & 15;
  int ny = n / SWS, nx = n % SWS;
  int py = (wy*SWS + ny + SSH) % SHW;
  int px = (wx*SWS + nx + SSH) % SHW;
  const float* src = x + (bb*12544 + py*SHW + px) * SC;
  float v0 = src[lane], v1 = src[lane+32], v2 = src[lane+64];
  float s = v0+v1+v2, sq = v0*v0 + v1*v1 + v2*v2;
  for (int o = 16; o >= 1; o >>= 1) {
    s  += __shfl_xor(s,  o, 32);
    sq += __shfl_xor(sq, o, 32);
  }
  float mean = s * (1.f/96.f);
  float var  = sq * (1.f/96.f) - mean*mean;
  float inv  = rsqrtf(var + 1e-5f);
  u16* dst = out + row * SC;
  dst[lane]    = f2bf((v0-mean)*inv*g[lane]    + bt[lane]);
  dst[lane+32] = f2bf((v1-mean)*inv*g[lane+32] + bt[lane+32]);
  dst[lane+64] = f2bf((v2-mean)*inv*g[lane+64] + bt[lane+64]);
}

// ---------------- K2: QKV GEMM, weights TDM-staged in LDS ----------------
__global__ __launch_bounds__(96) void k_qkv(const u16* __restrict__ xw,
                                            const u16* __restrict__ wt,
                                            const float* __restrict__ bias,
                                            u16* __restrict__ qb, u16* __restrict__ kb,
                                            u16* __restrict__ vb) {
  __shared__ u16 wsm[288*96];                    // 55,296 B weight panel
  if (threadIdx.x < 32)
    tdm_load_panel(wt, wsm, 288, 96);
  __syncthreads();

  int s = threadIdx.x >> 5;         // 0:q 1:k 2:v
  int lane = threadIdx.x & 31;
  int mt = blockIdx.x;
  int r = lane & 15, hf = lane >> 4;
  Frag a[3];
  const u16* arow = xw + (mt*16 + r) * SC;
  for (int ks = 0; ks < 3; ++ks)
    for (int v = 0; v < 8; ++v)
      a[ks].d[v] = *(const u32*)(arow + ks*32 + kofs(v, hf));
  u16* ob = (s == 0) ? qb : ((s == 1) ? kb : vb);
  float scale = (s == 0) ? SSCALE : 1.f;
  for (int nt = 0; nt < 6; ++nt) {
    const u16* brow = wsm + (s*96 + nt*16 + r) * SC;
    CAcc acc; acc.v = zero8();
    for (int ks = 0; ks < 3; ++ks) {
      Frag bf;
      for (int v = 0; v < 8; ++v)
        bf.d[v] = *(const u32*)(brow + ks*32 + kofs(v, hf));
      acc.v = wmma_bf16(a[ks], bf, acc.v);
    }
    int j  = nt*16 + r;             // col within section [0,96)
    int hh = j >> 5, dd = j & 31;
    float bj = bias[s*96 + j];
    for (int vg = 0; vg < 8; ++vg) {
      int row = mt*16 + vg + 8*hf;
      int w = row / SN, n = row % SN;
      ob[((w*3 + hh)*SN + n)*SHD + dd] = f2bf((acc.f[vg] + bj) * scale);
    }
  }
}

// ---------------- K3: windowed attention, one wave per (window, head) ----------------
__global__ __launch_bounds__(128) void k_attn(const u16* __restrict__ qb,
                                              const u16* __restrict__ kb,
                                              const u16* __restrict__ vb,
                                              const float* __restrict__ bias_attn,
                                              const float* __restrict__ amask,
                                              u16* __restrict__ ao) {
  __shared__ u16 plds[4][64*64];
  int wv = threadIdx.x >> 5, lane = threadIdx.x & 31;
  int gw = blockIdx.x * 4 + wv;     // (window*3 + head)
  int w = gw / 3, h = gw % 3;
  int r = lane & 15, hf = lane >> 4;
  const u16* qh = qb + gw * (SN*SHD);
  const u16* kh = kb + gw * (SN*SHD);
  const u16* vh = vb + gw * (SN*SHD);

  // Q fragments (rows padded 49 -> 64 with zeros)
  Frag qa[4];
  for (int mtm = 0; mtm < 4; ++mtm) {
    int row = mtm*16 + r;
    for (int v = 0; v < 8; ++v)
      qa[mtm].d[v] = (row < SN) ? *(const u32*)(qh + row*SHD + kofs(v, hf)) : 0u;
  }
  // S = q k^T   (HD=32 -> single K step)
  CAcc S[4][4];
  for (int ntk = 0; ntk < 4; ++ntk) {
    int key = ntk*16 + r;
    Frag kf;
    for (int v = 0; v < 8; ++v)
      kf.d[v] = (key < SN) ? *(const u32*)(kh + key*SHD + kofs(v, hf)) : 0u;
    for (int mtm = 0; mtm < 4; ++mtm)
      S[mtm][ntk].v = wmma_bf16(qa[mtm], kf, zero8());
  }
  // bias + mask + softmax (row reduce inside 16-lane half, matching C layout)
  const float* bh = bias_attn + h * 2401;
  const float* mh = amask + (w & 255) * 2401;
  for (int mtm = 0; mtm < 4; ++mtm) {
    for (int vg = 0; vg < 8; ++vg) {
      int m = mtm*16 + vg + 8*hf;
      float vals[4];
      for (int ntk = 0; ntk < 4; ++ntk) {
        int kc = ntk*16 + r;
        float sv = S[mtm][ntk].f[vg];
        sv = (m < SN && kc < SN) ? (sv + bh[m*SN + kc] + mh[m*SN + kc]) : -1e30f;
        vals[ntk] = sv;
      }
      float rmax = fmaxf(fmaxf(vals[0], vals[1]), fmaxf(vals[2], vals[3]));
      for (int o = 8; o >= 1; o >>= 1) rmax = fmaxf(rmax, __shfl_xor(rmax, o, 16));
      float rsum = 0.f;
      for (int ntk = 0; ntk < 4; ++ntk) { vals[ntk] = __expf(vals[ntk] - rmax); rsum += vals[ntk]; }
      for (int o = 8; o >= 1; o >>= 1) rsum += __shfl_xor(rsum, o, 16);
      float inv = 1.f / rsum;
      for (int ntk = 0; ntk < 4; ++ntk)
        plds[wv][m*64 + ntk*16 + r] = f2bf(vals[ntk] * inv);
    }
  }
  __syncthreads();
  // O = P v   (K = 64 padded keys -> 2 K steps; N = 32)
  CAcc O[4][2];
  for (int mtm = 0; mtm < 4; ++mtm)
    for (int ntv = 0; ntv < 2; ++ntv) O[mtm][ntv].v = zero8();
  for (int ks = 0; ks < 2; ++ks) {
    Frag vf[2];
    for (int ntv = 0; ntv < 2; ++ntv) {
      int dc = ntv*16 + r;
      for (int e = 0; e < 16; ++e) {
        int key = ks*32 + kelem(e, hf);
        vf[ntv].h[e] = (key < SN) ? vh[key*SHD + dc] : (u16)0;
      }
    }
    for (int mtm = 0; mtm < 4; ++mtm) {
      Frag pa;
      int row = mtm*16 + r;
      for (int v = 0; v < 8; ++v)
        pa.d[v] = *(const u32*)&plds[wv][row*64 + ks*32 + kofs(v, hf)];
      for (int ntv = 0; ntv < 2; ++ntv)
        O[mtm][ntv].v = wmma_bf16(pa, vf[ntv], O[mtm][ntv].v);
    }
  }
  for (int mtm = 0; mtm < 4; ++mtm)
    for (int ntv = 0; ntv < 2; ++ntv)
      for (int vg = 0; vg < 8; ++vg) {
        int m = mtm*16 + vg + 8*hf;
        if (m < SN)
          ao[(w*SN + m)*SC + h*SHD + ntv*16 + r] = f2bf(O[mtm][ntv].f[vg]);
      }
}

// ---------------- K4: proj GEMM (weights TDM-staged) + reverse + skip ----------------
__global__ __launch_bounds__(192) void k_proj(const u16* __restrict__ aoin,
                                              const u16* __restrict__ wt,
                                              const float* __restrict__ bias,
                                              const float* __restrict__ xin,
                                              float* __restrict__ xout) {
  __shared__ u16 wsm[96*96];                     // 18,432 B weight panel
  if (threadIdx.x < 32)
    tdm_load_panel(wt, wsm, 96, 96);
  __syncthreads();

  int t = threadIdx.x >> 5, lane = threadIdx.x & 31;
  int mt = blockIdx.x;
  int r = lane & 15, hf = lane >> 4;
  const u16* arow = aoin + (mt*16 + r) * SC;
  const u16* brow = wsm + (t*16 + r) * SC;
  CAcc acc; acc.v = zero8();
  for (int ks = 0; ks < 3; ++ks) {
    Frag a, b;
    for (int v = 0; v < 8; ++v) {
      a.d[v] = *(const u32*)(arow + ks*32 + kofs(v, hf));
      b.d[v] = *(const u32*)(brow + ks*32 + kofs(v, hf));
    }
    acc.v = wmma_bf16(a, b, acc.v);
  }
  int c = t*16 + r;
  float bc = bias[c];
  for (int vg = 0; vg < 8; ++vg) {
    int row = mt*16 + vg + 8*hf;
    int w = row / SN, n = row % SN;
    int bb = w >> 8, wi = w & 255, wy = wi >> 4, wx = wi & 15;
    int ny = n / SWS, nx = n % SWS;
    int py = (wy*SWS + ny + SSH) % SHW;
    int px = (wx*SWS + nx + SSH) % SHW;
    int tok = bb*12544 + py*SHW + px;
    xout[tok*SC + c] = xin[tok*SC + c] + acc.f[vg] + bc;
  }
}

// ---------------- K5: LN2 (natural token order) -> bf16 ----------------
__global__ __launch_bounds__(256) void k_ln2(const float* __restrict__ x,
                                             const float* __restrict__ g,
                                             const float* __restrict__ bt,
                                             u16* __restrict__ out) {
  int wv = threadIdx.x >> 5, lane = threadIdx.x & 31;
  int row = blockIdx.x * 8 + wv;
  const float* src = x + row * SC;
  float v0 = src[lane], v1 = src[lane+32], v2 = src[lane+64];
  float s = v0+v1+v2, sq = v0*v0 + v1*v1 + v2*v2;
  for (int o = 16; o >= 1; o >>= 1) {
    s  += __shfl_xor(s,  o, 32);
    sq += __shfl_xor(sq, o, 32);
  }
  float mean = s * (1.f/96.f);
  float var  = sq * (1.f/96.f) - mean*mean;
  float inv  = rsqrtf(var + 1e-5f);
  u16* dst = out + row * SC;
  dst[lane]    = f2bf((v0-mean)*inv*g[lane]    + bt[lane]);
  dst[lane+32] = f2bf((v1-mean)*inv*g[lane+32] + bt[lane+32]);
  dst[lane+64] = f2bf((v2-mean)*inv*g[lane+64] + bt[lane+64]);
}

// ---------------- K6: fc1 GEMM + exact GELU -> bf16 ----------------
__global__ __launch_bounds__(256) void k_fc1(const u16* __restrict__ xin,
                                             const u16* __restrict__ wt,
                                             const float* __restrict__ bias,
                                             u16* __restrict__ hout) {
  int t = threadIdx.x >> 5, lane = threadIdx.x & 31;
  int mt = blockIdx.x;
  int r = lane & 15, hf = lane >> 4;
  const u16* arow = xin + (mt*16 + r) * SC;
  Frag a[3];
  for (int ks = 0; ks < 3; ++ks)
    for (int v = 0; v < 8; ++v)
      a[ks].d[v] = *(const u32*)(arow + ks*32 + kofs(v, hf));
  for (int j = 0; j < 3; ++j) {
    int nt = t*3 + j;                 // 24 n-tiles across 8 waves
    const u16* brow = wt + (nt*16 + r) * SC;
    CAcc acc; acc.v = zero8();
    for (int ks = 0; ks < 3; ++ks) {
      Frag b;
      for (int v = 0; v < 8; ++v)
        b.d[v] = *(const u32*)(brow + ks*32 + kofs(v, hf));
      acc.v = wmma_bf16(a[ks], b, acc.v);
    }
    int c = nt*16 + r;
    float bc = bias[c];
    for (int vg = 0; vg < 8; ++vg) {
      int row = mt*16 + vg + 8*hf;
      float x = acc.f[vg] + bc;
      float ge = 0.5f * x * (1.f + erff(x * 0.70710678118654752f));
      hout[row*SHID + c] = f2bf(ge);
    }
  }
}

// ---------------- K7: fc2 GEMM + residual accumulate into d_out ----------------
__global__ __launch_bounds__(192) void k_fc2(const u16* __restrict__ hin,
                                             const u16* __restrict__ wt,
                                             const float* __restrict__ bias,
                                             float* __restrict__ xout) {
  int t = threadIdx.x >> 5, lane = threadIdx.x & 31;
  int mt = blockIdx.x;
  int r = lane & 15, hf = lane >> 4;
  const u16* arow = hin + (mt*16 + r) * SHID;
  const u16* brow = wt + (t*16 + r) * SHID;
  CAcc acc; acc.v = zero8();
  for (int ks = 0; ks < 12; ++ks) {
    Frag a, b;
    for (int v = 0; v < 8; ++v) {
      a.d[v] = *(const u32*)(arow + ks*32 + kofs(v, hf));
      b.d[v] = *(const u32*)(brow + ks*32 + kofs(v, hf));
    }
    acc.v = wmma_bf16(a, b, acc.v);
  }
  int c = t*16 + r;
  float bc = bias[c];
  for (int vg = 0; vg < 8; ++vg) {
    int row = mt*16 + vg + 8*hf;
    xout[row*SC + c] += acc.f[vg] + bc;
  }
}

// ---------------- launch ----------------
extern "C" void kernel_launch(void* const* d_in, const int* in_sizes, int n_in,
                              void* d_out, int out_size, void* d_ws, size_t ws_size,
                              hipStream_t stream) {
  const float* x      = (const float*)d_in[0];
  const float* g1     = (const float*)d_in[1];
  const float* b1     = (const float*)d_in[2];
  const float* qkv_w  = (const float*)d_in[3];
  const float* qkv_b  = (const float*)d_in[4];
  const float* proj_w = (const float*)d_in[5];
  const float* proj_b = (const float*)d_in[6];
  const float* rpb    = (const float*)d_in[7];
  const float* g2     = (const float*)d_in[8];
  const float* b2     = (const float*)d_in[9];
  const float* fc1_w  = (const float*)d_in[10];
  const float* fc1_b  = (const float*)d_in[11];
  const float* fc2_w  = (const float*)d_in[12];
  const float* fc2_b  = (const float*)d_in[13];
  const int*   rel    = (const int*)d_in[14];
  const float* amask  = (const float*)d_in[15];
  float* out = (float*)d_out;

  char* ws = (char*)d_ws;
  const size_t SZ_ACT = (size_t)SM0 * SC * 2;        // 154,140,672 B
  u16*   qkv_wt    = (u16*)(ws + 0);                 // 55296 B
  u16*   proj_wt   = (u16*)(ws + 55296);             // 18432 B
  u16*   fc1_wt    = (u16*)(ws + 73728);             // 73728 B
  u16*   fc2_wt    = (u16*)(ws + 147456);            // 73728 B
  float* bias_attn = (float*)(ws + 221184);          // 28812 B
  u16*   xw        = (u16*)(ws + 262144);            // LN out / LN2 out (reused)
  u16*   qb        = (u16*)(ws + 262144 + SZ_ACT);
  u16*   kb        = (u16*)(ws + 262144 + 2*SZ_ACT);
  u16*   vb        = (u16*)(ws + 262144 + 3*SZ_ACT);
  u16*   ao        = (u16*)(ws + 262144 + 4*SZ_ACT);
  u16*   hbuf      = qb;                              // fc1 out reuses q/k/v/ao (616 MB)

  k_prep<<<144, 256, 0, stream>>>(qkv_w, proj_w, fc1_w, fc2_w, rpb, rel,
                                  qkv_wt, proj_wt, fc1_wt, fc2_wt, bias_attn);
  k_ln1 <<<SM0/8, 256, 0, stream>>>(x, g1, b1, xw);
  k_qkv <<<SM0/16, 96, 0, stream>>>(xw, qkv_wt, qkv_b, qb, kb, vb);
  k_attn<<<(SWTOT*SNH)/4, 128, 0, stream>>>(qb, kb, vb, bias_attn, amask, ao);
  k_proj<<<SM0/16, 192, 0, stream>>>(ao, proj_wt, proj_b, x, out);
  k_ln2 <<<SM0/8, 256, 0, stream>>>(out, g2, b2, xw);
  k_fc1 <<<SM0/16, 256, 0, stream>>>(xw, fc1_wt, fc1_b, hbuf);
  k_fc2 <<<SM0/16, 192, 0, stream>>>(hbuf, fc2_wt, fc2_b, out);
}
